// SwinTransformerBlock_68272800137897
// MI455X (gfx1250) — matrix operand order repaired and checked by reading, not compiled, not run on hardware
//
#include <hip/hip_runtime.h>
#include <hip/hip_bf16.h>

// ---------------------------------------------------------------------------
// Swin Transformer block for MI455X (gfx1250, wave32, WMMA bf16)
// B=32, H=W=56, C=192, heads=6, hd=32, ws=7 (N=49), shift=3, nW=64
// ---------------------------------------------------------------------------

typedef __bf16 v16bf __attribute__((ext_vector_type(16)));
typedef __bf16 v8bf  __attribute__((ext_vector_type(8)));
typedef float  v8f   __attribute__((ext_vector_type(8)));

#define NHEADS 6
#define CDIM   192
#define HDIM   32
#define NTOK   100352     // 2048 windows * 49 tokens
#define HID    768
#define IMG    56
#define LTOK   3136       // 56*56

__device__ __forceinline__ v8f zero8() {
  v8f z = {0.f, 0.f, 0.f, 0.f, 0.f, 0.f, 0.f, 0.f};
  return z;
}

__device__ __forceinline__ float gelu_exact(float x) {
  return 0.5f * x * (1.0f + erff(x * 0.70710678118654752f));
}

// region label for the shifted-window attention mask (H-ws=49, H-shift=53)
__device__ __forceinline__ int region(int g) { return g < 49 ? 0 : (g < 53 ? 1 : 2); }

// Build a 16-element bf16 fragment from two 8-element (16B) loads.
__device__ __forceinline__ v16bf load_frag2(const __bf16* p0, const __bf16* p1) {
  v8bf a = *(const v8bf*)p0;
  v8bf b = *(const v8bf*)p1;
  v16bf r;
#pragma unroll
  for (int i = 0; i < 8; ++i) { r[i] = a[i]; r[i + 8] = b[i]; }
  return r;
}

// ---------------------------------------------------------------------------
// Wave-level GEMM core: one wave computes a 16(M) x 64(N) tile of
//   D = A(bf16, MxK row-major) * W^T (W is bf16 Nout x K row-major), fp32 acc.
// A-fragment layout (16x32 bf16): lanes 0-15 rows, K 0-15; lanes 16-31
// same rows, K 16-31  ->  two b128 loads per lane.
// B-fragment layout (32x16): lane n&15 = column; lanes<16 hold K 0-15,
// lanes>=16 hold K 16-31 -> 16 contiguous bf16 from W[n, k0+kb].
// ---------------------------------------------------------------------------
__device__ __forceinline__ void gemm_tile_16x64(const __bf16* __restrict__ A, int lda,
                                                const __bf16* __restrict__ W, int ldw,
                                                int m0, int n0, int K, v8f acc[4]) {
  const int lane = threadIdx.x & 31;
  const int arow = lane & 15;
  const int akb  = (lane >> 4) * 16;
  const int wn   = lane & 15;
  const int wkb  = (lane >> 4) * 16;
  const __bf16* ap = A + (size_t)(m0 + arow) * lda;
  for (int k0 = 0; k0 < K; k0 += 32) {
    v16bf a = load_frag2(ap + k0 + akb, ap + k0 + akb + 8);
#pragma unroll
    for (int t = 0; t < 4; ++t) {
      const __bf16* wp = W + (size_t)(n0 + 16 * t + wn) * ldw + k0 + wkb;
      v16bf b = load_frag2(wp, wp + 8);
      acc[t] = __builtin_amdgcn_wmma_f32_16x16x32_bf16(false, a, false, b,
                                                       (short)0, acc[t], false, false);
    }
  }
}

// ---------------------------------------------------------------------------
// Kernel 0: fp32 -> bf16 weight conversion
// ---------------------------------------------------------------------------
__global__ __launch_bounds__(256) void cvt_f32_bf16(const float* __restrict__ s,
                                                    __bf16* __restrict__ d, int n) {
  int i = blockIdx.x * blockDim.x + threadIdx.x;
  if (i < n) d[i] = (__bf16)s[i];
}

// ---------------------------------------------------------------------------
// Kernel 1: LN1 + cyclic shift (-3,-3) + window partition -> bf16 xw
// One wave per output window-token; 6 channels per lane; shuffle reductions.
// ---------------------------------------------------------------------------
__global__ __launch_bounds__(256) void ln1_win_kernel(const float* __restrict__ x,
                                                      const float* __restrict__ gw,
                                                      const float* __restrict__ gb,
                                                      __bf16* __restrict__ xw) {
  int tok  = blockIdx.x * 8 + (threadIdx.x >> 5);
  int lane = threadIdx.x & 31;
  if (tok >= NTOK) return;
  int w  = tok / 49, i = tok - w * 49;
  int b  = w >> 6, wi = w & 63;
  int sh = (wi >> 3) * 7 + i / 7;
  int sw = (wi & 7) * 7 + i % 7;
  int gh = sh + 3; if (gh >= IMG) gh -= IMG;   // roll(-3): read from (p+3)%56
  int gv = sw + 3; if (gv >= IMG) gv -= IMG;
  const float* row = x + ((size_t)b * LTOK + gh * IMG + gv) * CDIM;
  float v[6]; float s = 0.f, s2 = 0.f;
#pragma unroll
  for (int j = 0; j < 6; ++j) { float t = row[lane + 32 * j]; v[j] = t; s += t; s2 += t * t; }
#pragma unroll
  for (int o = 16; o > 0; o >>= 1) { s += __shfl_xor(s, o, 32); s2 += __shfl_xor(s2, o, 32); }
  float mu  = s * (1.f / 192.f);
  float var = s2 * (1.f / 192.f) - mu * mu;
  float rst = rsqrtf(var + 1e-5f);
  __bf16* dst = xw + (size_t)tok * CDIM;
#pragma unroll
  for (int j = 0; j < 6; ++j) {
    int c = lane + 32 * j;
    dst[c] = (__bf16)((v[j] - mu) * rst * gw[c] + gb[c]);
  }
}

// ---------------------------------------------------------------------------
// Kernel 2: QKV GEMM (100352 x 192) @ (192 x 576) + bias, scatter into
// per-(window,head) q/k/v bf16 buffers: [ (w*6+h)*49 + n ]*32 + d
// ---------------------------------------------------------------------------
__global__ __launch_bounds__(256) void qkv_gemm_kernel(const __bf16* __restrict__ A,
                                                       const __bf16* __restrict__ Wt,
                                                       const float* __restrict__ bias,
                                                       __bf16* __restrict__ q,
                                                       __bf16* __restrict__ k,
                                                       __bf16* __restrict__ v) {
  int wave = threadIdx.x >> 5, lane = threadIdx.x & 31;
  int m0 = blockIdx.x * 128 + wave * 16;
  int n0 = blockIdx.y * 64;
  v8f acc[4] = {zero8(), zero8(), zero8(), zero8()};
  gemm_tile_16x64(A, CDIM, Wt, CDIM, m0, n0, CDIM, acc);
  int colw = lane & 15, rowg = (lane >> 4) * 8;
  int wm[8], nm[8];
#pragma unroll
  for (int r = 0; r < 8; ++r) { int m = m0 + rowg + r; wm[r] = m / 49; nm[r] = m - wm[r] * 49; }
#pragma unroll
  for (int t = 0; t < 4; ++t) {
    int col = n0 + 16 * t + colw;
    float bia = bias[col];
    int s = col / CDIM; int rem = col - s * CDIM; int hh = rem >> 5; int d = rem & 31;
    __bf16* dst = (s == 0) ? q : (s == 1) ? k : v;
#pragma unroll
    for (int r = 0; r < 8; ++r) {
      dst[((size_t)(wm[r] * NHEADS + hh) * 49 + nm[r]) * HDIM + d] = (__bf16)(acc[t][r] + bia);
    }
  }
}

// ---------------------------------------------------------------------------
// Kernel 3: windowed attention, one block (128 thr / 4 waves) per (window,head)
// S = q k^T * scale + rel_bias + shift_mask ; softmax ; O = P v
// Padded to 64x64 in LDS; rel-pos index and mask computed arithmetically.
// ---------------------------------------------------------------------------
__global__ __launch_bounds__(128) void attn_kernel(const __bf16* __restrict__ qg,
                                                   const __bf16* __restrict__ kg,
                                                   const __bf16* __restrict__ vg,
                                                   const float* __restrict__ btab,
                                                   __bf16* __restrict__ aout) {
  __shared__ __bf16 qs[64 * 32];
  __shared__ __bf16 kls[64 * 32];
  __shared__ __bf16 vts[32 * 64];   // v transposed: [d][key]
  __shared__ __bf16 ps[64 * 64];    // softmax probs, bf16

  const int wh  = blockIdx.x;             // window*NHEADS + head
  const int w   = wh / NHEADS;
  const int h   = wh - w * NHEADS;
  const int wi  = w & 63;
  const int wh7 = wi >> 3, ww7 = wi & 7;
  const size_t base = (size_t)wh * 49 * HDIM;
  const int tid = threadIdx.x;
  const __bf16 zbf = (__bf16)0.f;

  for (int idx = tid; idx < 2048; idx += 128) {
    int key = idx >> 5, d = idx & 31;
    bool ok = key < 49;
    qs[idx]         = ok ? qg[base + idx] : zbf;
    kls[idx]        = ok ? kg[base + idx] : zbf;
    vts[d * 64 + key] = ok ? vg[base + idx] : zbf;
  }
  __syncthreads();

  const int wave = tid >> 5, lane = tid & 31;
  const int arow = lane & 15, akb = (lane >> 4) * 16;
  const int colw = lane & 15, wkb = (lane >> 4) * 16;
  const int rowbase = wave * 16 + (lane >> 4) * 8;

  // ---- S = q @ k^T (K=32, one WMMA per 16x16 tile, 4 tiles per wave) ----
  const __bf16* qp = &qs[(wave * 16 + arow) * 32 + akb];
  v16bf aq = load_frag2(qp, qp + 8);
  v8f acc[4] = {zero8(), zero8(), zero8(), zero8()};
#pragma unroll
  for (int t = 0; t < 4; ++t) {
    const __bf16* kp = &kls[(16 * t + colw) * 32 + wkb];
    v16bf bk = load_frag2(kp, kp + 8);
    acc[t] = __builtin_amdgcn_wmma_f32_16x16x32_bf16(false, aq, false, bk,
                                                     (short)0, acc[t], false, false);
  }

  // ---- bias + mask + padding ----
  int jy[4], jx[4], jl[4]; bool jv[4];
#pragma unroll
  for (int t = 0; t < 4; ++t) {
    int j = 16 * t + colw;
    jv[t] = j < 49;
    int jc = jv[t] ? j : 48;
    jy[t] = jc / 7; jx[t] = jc - jy[t] * 7;
    jl[t] = region(wh7 * 7 + jy[t]) * 3 + region(ww7 * 7 + jx[t]);
  }
  const float scale = 0.17677669529663687f;   // 32^-0.5
  float sc[4][8];
#pragma unroll
  for (int r = 0; r < 8; ++r) {
    int m  = rowbase + r;
    int mc = m < 49 ? m : 48;
    int yi = mc / 7, xi = mc - (mc / 7) * 7;
    int li = region(wh7 * 7 + yi) * 3 + region(ww7 * 7 + xi);
#pragma unroll
    for (int t = 0; t < 4; ++t) {
      int idx = (yi - jy[t] + 6) * 13 + (xi - jx[t] + 6);
      float s = acc[t][r] * scale + btab[idx * NHEADS + h];
      if (li != jl[t]) s -= 100.f;
      if (!jv[t]) s = -1e30f;
      sc[t][r] = s;
    }
  }

  // ---- row softmax: rows live in one 16-lane half -> xor-butterfly 1,2,4,8 ----
#pragma unroll
  for (int r = 0; r < 8; ++r) {
    float mx = fmaxf(fmaxf(sc[0][r], sc[1][r]), fmaxf(sc[2][r], sc[3][r]));
#pragma unroll
    for (int o = 1; o < 16; o <<= 1) mx = fmaxf(mx, __shfl_xor(mx, o, 32));
    float e[4], sum = 0.f;
#pragma unroll
    for (int t = 0; t < 4; ++t) { e[t] = __expf(sc[t][r] - mx); sum += e[t]; }
#pragma unroll
    for (int o = 1; o < 16; o <<= 1) sum += __shfl_xor(sum, o, 32);
    float inv = 1.f / sum;
#pragma unroll
    for (int t = 0; t < 4; ++t) ps[(rowbase + r) * 64 + 16 * t + colw] = (__bf16)(e[t] * inv);
  }
  __syncthreads();

  // ---- O = P @ v : K=64 (2 steps), 2 output tiles (32 head dims) ----
  v8f o0 = zero8(), o1 = zero8();
#pragma unroll
  for (int kk = 0; kk < 64; kk += 32) {
    const __bf16* pp = &ps[(wave * 16 + arow) * 64 + kk + akb];
    v16bf aP = load_frag2(pp, pp + 8);
    const __bf16* v0p = &vts[(0 + colw) * 64 + kk + wkb];
    const __bf16* v1p = &vts[(16 + colw) * 64 + kk + wkb];
    v16bf b0 = load_frag2(v0p, v0p + 8);
    v16bf b1 = load_frag2(v1p, v1p + 8);
    o0 = __builtin_amdgcn_wmma_f32_16x16x32_bf16(false, aP, false, b0, (short)0, o0, false, false);
    o1 = __builtin_amdgcn_wmma_f32_16x16x32_bf16(false, aP, false, b1, (short)0, o1, false, false);
  }
#pragma unroll
  for (int r = 0; r < 8; ++r) {
    int m = rowbase + r;
    if (m < 49) {
      size_t ob = ((size_t)w * 49 + m) * CDIM + h * HDIM;
      aout[ob + colw]      = (__bf16)o0[r];
      aout[ob + 16 + colw] = (__bf16)o1[r];
    }
  }
}

// ---------------------------------------------------------------------------
// Kernel 4: proj GEMM + window-reverse + roll(+3) + residual -> fp32 x1
// ---------------------------------------------------------------------------
__global__ __launch_bounds__(256) void proj_gemm_kernel(const __bf16* __restrict__ A,
                                                        const __bf16* __restrict__ Wt,
                                                        const float* __restrict__ bias,
                                                        const float* __restrict__ xin,
                                                        float* __restrict__ x1) {
  int wave = threadIdx.x >> 5, lane = threadIdx.x & 31;
  int m0 = blockIdx.x * 128 + wave * 16;
  int n0 = blockIdx.y * 64;
  v8f acc[4] = {zero8(), zero8(), zero8(), zero8()};
  gemm_tile_16x64(A, CDIM, Wt, CDIM, m0, n0, CDIM, acc);
  int colw = lane & 15, rowg = (lane >> 4) * 8;
  size_t tokoff[8];
#pragma unroll
  for (int r = 0; r < 8; ++r) {
    int m = m0 + rowg + r;
    int w = m / 49, i = m - w * 49;
    int b = w >> 6, wi = w & 63;
    int sh = (wi >> 3) * 7 + i / 7;
    int sw = (wi & 7) * 7 + i % 7;
    int gh = sh + 3; if (gh >= IMG) gh -= IMG;   // reverse roll(+3)
    int gv = sw + 3; if (gv >= IMG) gv -= IMG;
    tokoff[r] = ((size_t)b * LTOK + gh * IMG + gv) * CDIM;
  }
#pragma unroll
  for (int t = 0; t < 4; ++t) {
    int col = n0 + 16 * t + colw;
    float bia = bias[col];
#pragma unroll
    for (int r = 0; r < 8; ++r) {
      x1[tokoff[r] + col] = xin[tokoff[r] + col] + acc[t][r] + bia;
    }
  }
}

// ---------------------------------------------------------------------------
// Kernel 5: LN2 (natural token order) -> bf16
// ---------------------------------------------------------------------------
__global__ __launch_bounds__(256) void ln2_kernel(const float* __restrict__ x1,
                                                  const float* __restrict__ gw,
                                                  const float* __restrict__ gb,
                                                  __bf16* __restrict__ h2) {
  int tok  = blockIdx.x * 8 + (threadIdx.x >> 5);
  int lane = threadIdx.x & 31;
  if (tok >= NTOK) return;
  const float* row = x1 + (size_t)tok * CDIM;
  float v[6]; float s = 0.f, s2 = 0.f;
#pragma unroll
  for (int j = 0; j < 6; ++j) { float t = row[lane + 32 * j]; v[j] = t; s += t; s2 += t * t; }
#pragma unroll
  for (int o = 16; o > 0; o >>= 1) { s += __shfl_xor(s, o, 32); s2 += __shfl_xor(s2, o, 32); }
  float mu  = s * (1.f / 192.f);
  float var = s2 * (1.f / 192.f) - mu * mu;
  float rst = rsqrtf(var + 1e-5f);
  __bf16* dst = h2 + (size_t)tok * CDIM;
#pragma unroll
  for (int j = 0; j < 6; ++j) {
    int c = lane + 32 * j;
    dst[c] = (__bf16)((v[j] - mu) * rst * gw[c] + gb[c]);
  }
}

// ---------------------------------------------------------------------------
// Kernel 6: fc1 GEMM + exact GELU -> bf16
// ---------------------------------------------------------------------------
__global__ __launch_bounds__(256) void fc1_gemm_kernel(const __bf16* __restrict__ A,
                                                       const __bf16* __restrict__ Wt,
                                                       const float* __restrict__ bias,
                                                       __bf16* __restrict__ m1) {
  int wave = threadIdx.x >> 5, lane = threadIdx.x & 31;
  int m0 = blockIdx.x * 128 + wave * 16;
  int n0 = blockIdx.y * 64;
  v8f acc[4] = {zero8(), zero8(), zero8(), zero8()};
  gemm_tile_16x64(A, CDIM, Wt, CDIM, m0, n0, CDIM, acc);
  int colw = lane & 15, rowg = (lane >> 4) * 8;
#pragma unroll
  for (int t = 0; t < 4; ++t) {
    int col = n0 + 16 * t + colw;
    float bia = bias[col];
#pragma unroll
    for (int r = 0; r < 8; ++r) {
      int m = m0 + rowg + r;
      m1[(size_t)m * HID + col] = (__bf16)gelu_exact(acc[t][r] + bia);
    }
  }
}

// ---------------------------------------------------------------------------
// Kernel 7: fc2 GEMM + residual -> fp32 output
// ---------------------------------------------------------------------------
__global__ __launch_bounds__(256) void fc2_gemm_kernel(const __bf16* __restrict__ A,
                                                       const __bf16* __restrict__ Wt,
                                                       const float* __restrict__ bias,
                                                       const float* __restrict__ x1,
                                                       float* __restrict__ out) {
  int wave = threadIdx.x >> 5, lane = threadIdx.x & 31;
  int m0 = blockIdx.x * 128 + wave * 16;
  int n0 = blockIdx.y * 64;
  v8f acc[4] = {zero8(), zero8(), zero8(), zero8()};
  gemm_tile_16x64(A, HID, Wt, HID, m0, n0, HID, acc);
  int colw = lane & 15, rowg = (lane >> 4) * 8;
#pragma unroll
  for (int t = 0; t < 4; ++t) {
    int col = n0 + 16 * t + colw;
    float bia = bias[col];
#pragma unroll
    for (int r = 0; r < 8; ++r) {
      size_t m = (size_t)(m0 + rowg + r);
      out[m * CDIM + col] = x1[m * CDIM + col] + acc[t][r] + bia;
    }
  }
}

// ---------------------------------------------------------------------------
// Host launcher
// ---------------------------------------------------------------------------
extern "C" void kernel_launch(void* const* d_in, const int* in_sizes, int n_in,
                              void* d_out, int out_size, void* d_ws, size_t ws_size,
                              hipStream_t stream) {
  (void)in_sizes; (void)n_in; (void)out_size; (void)ws_size;
  const float* x      = (const float*)d_in[0];
  const float* ln1_w  = (const float*)d_in[1];
  const float* ln1_b  = (const float*)d_in[2];
  const float* qkv_w  = (const float*)d_in[3];
  const float* qkv_b  = (const float*)d_in[4];
  const float* proj_w = (const float*)d_in[5];
  const float* proj_b = (const float*)d_in[6];
  const float* btab   = (const float*)d_in[7];
  const float* ln2_w  = (const float*)d_in[8];
  const float* ln2_b  = (const float*)d_in[9];
  const float* fc1_w  = (const float*)d_in[10];
  const float* fc1_b  = (const float*)d_in[11];
  const float* fc2_w  = (const float*)d_in[12];
  const float* fc2_b  = (const float*)d_in[13];
  float* out = (float*)d_out;

  char* base = (char*)d_ws;
  size_t off = 0;
  auto take = [&](size_t bytes) -> char* {
    char* r = base + off;
    off = (off + bytes + 255) & ~(size_t)255;
    return r;
  };
  const size_t NE = (size_t)NTOK * CDIM;          // 19,267,584 elements
  __bf16* wqkv = (__bf16*)take(576 * 192 * sizeof(__bf16));
  __bf16* wprj = (__bf16*)take(192 * 192 * sizeof(__bf16));
  __bf16* wfc1 = (__bf16*)take(768 * 192 * sizeof(__bf16));
  __bf16* wfc2 = (__bf16*)take(192 * 768 * sizeof(__bf16));
  __bf16* xw   = (__bf16*)take(NE * sizeof(__bf16));   // also reused as attn_out
  __bf16* qb   = (__bf16*)take(NE * sizeof(__bf16));   // also reused as h2
  __bf16* kb   = (__bf16*)take(NE * sizeof(__bf16));
  __bf16* vb   = (__bf16*)take(NE * sizeof(__bf16));
  float*  x1   = (float*)take(NE * sizeof(float));
  __bf16* m1   = (__bf16*)take((size_t)NTOK * HID * sizeof(__bf16));

  // 0) weight conversion
  cvt_f32_bf16<<<(576 * 192 + 255) / 256, 256, 0, stream>>>(qkv_w, wqkv, 576 * 192);
  cvt_f32_bf16<<<(192 * 192 + 255) / 256, 256, 0, stream>>>(proj_w, wprj, 192 * 192);
  cvt_f32_bf16<<<(768 * 192 + 255) / 256, 256, 0, stream>>>(fc1_w, wfc1, 768 * 192);
  cvt_f32_bf16<<<(192 * 768 + 255) / 256, 256, 0, stream>>>(fc2_w, wfc2, 192 * 768);

  // 1) LN1 + shift + window partition
  ln1_win_kernel<<<NTOK / 8, 256, 0, stream>>>(x, ln1_w, ln1_b, xw);

  // 2) QKV GEMM (M=100352, N=576, K=192)
  qkv_gemm_kernel<<<dim3(NTOK / 128, 576 / 64), 256, 0, stream>>>(xw, wqkv, qkv_b, qb, kb, vb);

  // 3) windowed attention (2048 windows * 6 heads); reuse xw as attn_out
  attn_kernel<<<2048 * NHEADS, 128, 0, stream>>>(qb, kb, vb, btab, xw);

  // 4) proj GEMM + reverse + roll + residual (N=192)
  proj_gemm_kernel<<<dim3(NTOK / 128, 192 / 64), 256, 0, stream>>>(xw, wprj, proj_b, x, x1);

  // 5) LN2 (reuse qb as h2)
  ln2_kernel<<<NTOK / 8, 256, 0, stream>>>(x1, ln2_w, ln2_b, qb);

  // 6) fc1 GEMM + GELU (N=768)
  fc1_gemm_kernel<<<dim3(NTOK / 128, 768 / 64), 256, 0, stream>>>(qb, wfc1, fc1_b, m1);

  // 7) fc2 GEMM + residual (K=768, N=192) -> output
  fc2_gemm_kernel<<<dim3(NTOK / 128, 192 / 64), 256, 0, stream>>>(m1, wfc2, fc2_b, x1, out);
}
